// EquivariantConvolution_73229192397395
// MI455X (gfx1250) — matrix operand order repaired and checked by reading, not compiled, not run on hardware
//
#include <hip/hip_runtime.h>
#include <math.h>

typedef __attribute__((ext_vector_type(16))) _Float16 v16h;
typedef __attribute__((ext_vector_type(8)))  float    v8f;

#define LMAX    2
#define MUL_IN  16
#define OUT_CH  16
#define NUM_RB  11
#define RAD_HID 12
#define NPATH   11
#define WCOLS   2816      // 11 * 16 * 16
#define FROW    144       // MUL_IN * 9
#define CGPAD   125       // 5*5*5 padded per path
// workspace layout (bytes):
//   [0 .. 2817*16*2)            : w2h  f16, [col][16] rows 0..11 + zero pad,
//                                 col==2816 is an all-zero row (for hi lanes)
//   [W2H_BYTES .. +CG bytes)    : cg   f32, NPATH * 125
#define W2H_BYTES ((WCOLS + 1) * 16 * 2)

// ---------------------------------------------------------------------------
// Path table (must match Python loop order: l1 outer, l2, l3 inner)
// ---------------------------------------------------------------------------
static __device__ const int PL1[NPATH] = {0,0,0,1,1,1,1,2,2,2,2};
static __device__ const int PL2[NPATH] = {0,1,2,0,1,1,2,0,1,2,2};
static __device__ const int PL3[NPATH] = {0,1,2,1,0,2,1,2,1,0,2};

// ---------------------------------------------------------------------------
// Kernel 0: zero the output (segment_sum accumulator)
// ---------------------------------------------------------------------------
__global__ void zero_out_kernel(float* out, int n) {
    int i = blockIdx.x * blockDim.x + threadIdx.x;
    if (i < n) out[i] = 0.0f;
}

// ---------------------------------------------------------------------------
// Kernel 1: convert W2 (12 x 2816 f32, row-major) into contiguous-k f16
// layout: w2h[col][r] for r=0..15 (rows 12..15 zero), col = 0..2816 where
// col 2816 is an all-zero row used by lanes 16..31 (K=16..31 half of B).
// ---------------------------------------------------------------------------
__global__ void w2cvt_kernel(const float* __restrict__ W2,
                             _Float16* __restrict__ w2h) {
    int idx = blockIdx.x * blockDim.x + threadIdx.x;
    int total = (WCOLS + 1) * 16;
    if (idx >= total) return;
    int col = idx >> 4;
    int r   = idx & 15;
    float v = (col < WCOLS && r < RAD_HID) ? W2[r * WCOLS + col] : 0.0f;
    w2h[idx] = (_Float16)v;
}

// ---------------------------------------------------------------------------
// Kernel 2: compute normalized CG tensors via 10x24 Gauss-Legendre x phi
// quadrature (faithful to the reference), store padded [5][5][5] per path.
// ---------------------------------------------------------------------------
__device__ double shY(int l, int m, double x, double y, double z) {
    const double s3  = 1.7320508075688772;
    const double s5  = 2.23606797749979;
    const double s15 = 3.872983346207417;
    if (l == 0) return 1.0;
    if (l == 1) return (m == 0) ? s3 * y : ((m == 1) ? s3 * z : s3 * x);
    switch (m) {
        case 0:  return s15 * x * y;
        case 1:  return s15 * y * z;
        case 2:  return 0.5 * s5 * (3.0 * z * z - 1.0);
        case 3:  return s15 * x * z;
        default: return 0.5 * s15 * (x * x - y * y);
    }
}

__global__ void cg_init_kernel(float* cg) {
    __shared__ float g[CGPAD];
    __shared__ float inv_norm;

    const double gl_t[10] = {
        -0.9739065285171717, -0.8650633666889845, -0.6794095682990244,
        -0.4333953941292472, -0.1488743389816312,  0.1488743389816312,
         0.4333953941292472,  0.6794095682990244,  0.8650633666889845,
         0.9739065285171717 };
    const double gl_w[10] = {
        0.06667134430868814, 0.14945134915058059, 0.21908636251598204,
        0.26926671930999635, 0.29552422471475287, 0.29552422471475287,
        0.26926671930999635, 0.21908636251598204, 0.14945134915058059,
        0.06667134430868814 };

    int p  = blockIdx.x;
    int l1 = PL1[p], l2 = PL2[p], l3 = PL3[p];
    int D1 = 2 * l1 + 1, D2 = 2 * l2 + 1, D3 = 2 * l3 + 1;
    int n  = D1 * D2 * D3;

    for (int idx = threadIdx.x; idx < CGPAD; idx += blockDim.x) g[idx] = 0.0f;
    __syncthreads();

    const double dphi = 6.283185307179586476925286766559 / 24.0;
    for (int idx = threadIdx.x; idx < n; idx += blockDim.x) {
        int i = idx / (D2 * D3);
        int r = idx % (D2 * D3);
        int j = r / D3;
        int k = r % D3;
        double sum = 0.0;
        for (int a = 0; a < 10; ++a) {
            double t  = gl_t[a];
            double st = sqrt(1.0 - t * t);
            double wq = gl_w[a] / 48.0;   // wt * (2pi/24) / (4pi)
            for (int b = 0; b < 24; ++b) {
                double ph = b * dphi;
                double x = st * cos(ph);
                double y = st * sin(ph);
                double z = t;
                sum += shY(l1, i, x, y, z) * shY(l2, j, x, y, z) *
                       shY(l3, k, x, y, z) * wq;
            }
        }
        g[i * 25 + j * 5 + k] = (float)sum;
    }
    __syncthreads();
    if (threadIdx.x == 0) {
        float ss = 0.0f;
        for (int t = 0; t < CGPAD; ++t) ss += g[t] * g[t];
        inv_norm = 1.0f / sqrtf(ss);
    }
    __syncthreads();
    float s = inv_norm;
    for (int idx = threadIdx.x; idx < CGPAD; idx += blockDim.x)
        cg[p * CGPAD + idx] = g[idx] * s;
}

// ---------------------------------------------------------------------------
// Kernel 3: fused equivariant convolution. One wave32 per 16-edge tile.
//   Phase A : radial basis -> h (scales folded), spherical harmonics
//   Per path: tmp[e,u,k] = alpha * sum_{i,j} f[e,u,i] sh[e,j] CG[i,j,k] (LDS)
//             wb tile via v_wmma_f32_16x16x32_f16 : h(16x12) @ W2(12x16),
//             B fragment = one contiguous 32B load from pre-laid-out w2h
//             acc[e,w,k] += tmp[e,u,k] * wb[e,u,w]   (D-layout register accum)
//   Epilogue: atomicAdd scatter to out[src].
// u-loop unrolled x4 (4 B-loads + 4 WMMAs in flight); no occupancy clamp so
// the allocator takes its natural ~200-250 VGPRs without spilling.
// ---------------------------------------------------------------------------
__global__ __launch_bounds__(32)
void equiv_conv_kernel(const float*    __restrict__ pos,
                       const float*    __restrict__ feat,
                       const int*      __restrict__ ei,
                       const float*    __restrict__ W1,
                       const _Float16* __restrict__ w2h,
                       const float*    __restrict__ cg,
                       float*          __restrict__ out,
                       int E) {
    __shared__ float h_lds[16][RAD_HID];
    __shared__ float sh_lds[16][9];
    __shared__ float tmp_lds[16][16][5];
    __shared__ int   src_lds[16];
    __shared__ int   dst_lds[16];

    const int lane = threadIdx.x;
    const int m    = lane & 15;       // WMMA row (edge) / column (w) lane id
    const int hi   = lane >> 4;       // lane half
    const int e0   = blockIdx.x * 16;

    // ---- Phase A: per-edge radial MLP + spherical harmonics (lanes 0..15)
    if (lane < 16) {
        int e = e0 + lane;
        if (e < E) {
            int s = ei[e];
            int d = ei[E + e];
            src_lds[lane] = s;
            dst_lds[lane] = d;
            float vx = pos[3 * s + 0] - pos[3 * d + 0];
            float vy = pos[3 * s + 1] - pos[3 * d + 1];
            float vz = pos[3 * s + 2] - pos[3 * d + 2];
            float dist = sqrtf(vx * vx + vy * vy + vz * vz + 1e-12f);

            float rb[NUM_RB];
            #pragma unroll
            for (int i = 0; i < NUM_RB; ++i) {
                float dd = (dist - 0.8f * i) * 1.25f;
                rb[i] = expf(-dd * dd) * (1.0f / 1.12f);
            }
            const float s1 = 0.30151134457776363f;  // 1/sqrt(11)
            const float s2 = 0.28867513459481287f;  // 1/sqrt(12), folded into h
            #pragma unroll
            for (int j = 0; j < RAD_HID; ++j) {
                float a = 0.0f;
                #pragma unroll
                for (int i = 0; i < NUM_RB; ++i)
                    a = fmaf(rb[i], W1[i * RAD_HID + j], a);
                a *= s1;
                h_lds[lane][j] = (a > 0.0f ? a : 0.0f) * s2;
            }
            float inv = 1.0f / dist;
            float x = vx * inv, y = vy * inv, z = vz * inv;
            const float s3f = 1.7320508075688772f;
            const float s5f = 2.23606797749979f;
            const float s15f = 3.872983346207417f;
            sh_lds[lane][0] = 1.0f;
            sh_lds[lane][1] = s3f * y;
            sh_lds[lane][2] = s3f * z;
            sh_lds[lane][3] = s3f * x;
            sh_lds[lane][4] = s15f * x * y;
            sh_lds[lane][5] = s15f * y * z;
            sh_lds[lane][6] = 0.5f * s5f * (3.0f * z * z - 1.0f);
            sh_lds[lane][7] = s15f * x * z;
            sh_lds[lane][8] = 0.5f * s15f * (x * x - y * y);
        } else {
            src_lds[lane] = 0;
            dst_lds[lane] = 0;
            #pragma unroll
            for (int j = 0; j < RAD_HID; ++j) h_lds[lane][j] = 0.0f;
            #pragma unroll
            for (int j = 0; j < 9; ++j) sh_lds[lane][j] = 0.0f;
        }
    }
    __syncthreads();

    // ---- A fragment: 16x32 f16, K=12 valid (contiguous k = 16*hi + s)
    v16h afrag;
    #pragma unroll
    for (int v = 0; v < 8; ++v) {
        int K0 = 2 * v + (hi ? 16 : 0);
        afrag[2 * v]     = (K0     < RAD_HID) ? (_Float16)h_lds[m][K0]     : (_Float16)0.0f;
        afrag[2 * v + 1] = (K0 + 1 < RAD_HID) ? (_Float16)h_lds[m][K0 + 1] : (_Float16)0.0f;
    }

    // Per-lane base into w2h: lanes 0..15 read real columns, lanes 16..31
    // read the all-zero row at col==WCOLS (K=16..31 half of B must be zero).
    const _Float16* w2base = w2h + ((hi == 0) ? (size_t)m : (size_t)WCOLS) * 16;
    const int       w2step = (hi == 0) ? 16 : 0;   // advance only for lo lanes

    float acc[8][9];
    #pragma unroll
    for (int r = 0; r < 8; ++r)
        #pragma unroll
        for (int k = 0; k < 9; ++k) acc[r][k] = 0.0f;

#define PATH_BODY(P, L1, L2, L3, ALPHA)                                        \
    {                                                                          \
        constexpr int D1 = 2 * (L1) + 1;                                       \
        constexpr int D2 = 2 * (L2) + 1;                                       \
        constexpr int D3 = 2 * (L3) + 1;                                       \
        constexpr int FOFF = ((L1) == 0) ? 0 : (((L1) == 1) ? 16 : 64);        \
        constexpr int SOFF = ((L2) == 0) ? 0 : (((L2) == 1) ? 1 : 4);          \
        constexpr int KOFF = ((L3) == 0) ? 0 : (((L3) == 1) ? 1 : 4);          \
        const float* cgp = cg + (P) * CGPAD;                                   \
        __syncthreads(); /* previous path's tmp consumption done */            \
        _Pragma("unroll 2")                                                    \
        for (int s8 = 0; s8 < 8; ++s8) {                                       \
            int idx = lane + 32 * s8;                                          \
            int te = idx >> 4, tu = idx & 15;                                  \
            float t[D3];                                                       \
            _Pragma("unroll") for (int k = 0; k < D3; ++k) t[k] = 0.0f;        \
            int db = dst_lds[te] * FROW + FOFF + tu * D1;                      \
            _Pragma("unroll") for (int i = 0; i < D1; ++i) {                   \
                float fi = feat[db + i];                                       \
                _Pragma("unroll") for (int j = 0; j < D2; ++j) {               \
                    float fs = fi * sh_lds[te][SOFF + j];                      \
                    _Pragma("unroll") for (int k = 0; k < D3; ++k)             \
                        t[k] = fmaf(fs, cgp[i * 25 + j * 5 + k], t[k]);        \
                }                                                              \
            }                                                                  \
            _Pragma("unroll") for (int k = 0; k < D3; ++k)                     \
                tmp_lds[te][tu][k] = (ALPHA) * t[k];                           \
        }                                                                      \
        __syncthreads(); /* tmp ready */                                       \
        _Pragma("unroll 4")                                                    \
        for (int u = 0; u < 16; ++u) {                                         \
            /* one contiguous 32-byte f16 load = full B fragment */            \
            v16h bfrag = *(const v16h*)(w2base +                               \
                                        (size_t)((P) * 256 + u * 16) * w2step);\
            v8f cz = {0.f, 0.f, 0.f, 0.f, 0.f, 0.f, 0.f, 0.f};                 \
            v8f dm = __builtin_amdgcn_wmma_f32_16x16x32_f16(                   \
                false, afrag, false, bfrag, (short)0, cz, false, false);       \
            _Pragma("unroll") for (int r = 0; r < 8; ++r) {                    \
                float wb = dm[r];                                              \
                int el = r + hi * 8;                                           \
                _Pragma("unroll") for (int k = 0; k < D3; ++k)                 \
                    acc[r][KOFF + k] =                                         \
                        fmaf(tmp_lds[el][u][k], wb, acc[r][KOFF + k]);         \
            }                                                                  \
        }                                                                      \
    }

    PATH_BODY(0, 0, 0, 0, 0.14433756729740646f)
    PATH_BODY(1, 0, 1, 1, 0.125f)
    PATH_BODY(2, 0, 2, 2, 0.125f)
    PATH_BODY(3, 1, 0, 1, 0.125f)
    PATH_BODY(4, 1, 1, 0, 0.14433756729740646f)
    PATH_BODY(5, 1, 1, 2, 0.125f)
    PATH_BODY(6, 1, 2, 1, 0.125f)
    PATH_BODY(7, 2, 0, 2, 0.125f)
    PATH_BODY(8, 2, 1, 1, 0.125f)
    PATH_BODY(9, 2, 2, 0, 0.14433756729740646f)
    PATH_BODY(10, 2, 2, 2, 0.125f)
#undef PATH_BODY

    // ---- Epilogue: scatter-add into out[src] (segment_sum)
    #pragma unroll
    for (int r = 0; r < 8; ++r) {
        int el = r + hi * 8;
        int e  = e0 + el;
        if (e < E) {
            int base = src_lds[el] * FROW;
            int w = m;
            atomicAdd(out + base + 0 + w, acc[r][0]);
            #pragma unroll
            for (int k = 0; k < 3; ++k)
                atomicAdd(out + base + 16 + w * 3 + k, acc[r][1 + k]);
            #pragma unroll
            for (int k = 0; k < 5; ++k)
                atomicAdd(out + base + 64 + w * 5 + k, acc[r][4 + k]);
        }
    }
}

// ---------------------------------------------------------------------------
extern "C" void kernel_launch(void* const* d_in, const int* in_sizes, int n_in,
                              void* d_out, int out_size, void* d_ws, size_t ws_size,
                              hipStream_t stream) {
    const float* pos  = (const float*)d_in[0];
    const float* feat = (const float*)d_in[1];
    const int*   ei   = (const int*)d_in[2];
    const float* W1   = (const float*)d_in[3];
    const float* W2   = (const float*)d_in[4];
    float* out = (float*)d_out;

    _Float16* w2h = (_Float16*)d_ws;                        // 90 KB
    float*    cg  = (float*)((char*)d_ws + W2H_BYTES);      // 5.5 KB

    int E = in_sizes[2] / 2;

    zero_out_kernel<<<(out_size + 255) / 256, 256, 0, stream>>>(out, out_size);
    w2cvt_kernel<<<((WCOLS + 1) * 16 + 255) / 256, 256, 0, stream>>>(W2, w2h);
    cg_init_kernel<<<NPATH, 128, 0, stream>>>(cg);
    equiv_conv_kernel<<<(E + 15) / 16, 32, 0, stream>>>(pos, feat, ei, W1, w2h,
                                                        cg, out, E);
}
